// GatedSelfAttention_72533407695350
// MI455X (gfx1250) — matrix operand order repaired and checked by reading, not compiled, not run on hardware
//
#include <hip/hip_runtime.h>
#include <hip/hip_bf16.h>

typedef __bf16 bf16_t;
typedef __attribute__((ext_vector_type(16))) __bf16 v16bf;
typedef __attribute__((ext_vector_type(8)))  __bf16 v8bf;
typedef __attribute__((ext_vector_type(8)))  float  v8f;
typedef __attribute__((ext_vector_type(4)))  int    v4i;

typedef __attribute__((address_space(1))) v4i gv4i_t;  // global int4
typedef __attribute__((address_space(3))) v4i lv4i_t;  // LDS int4

#define NB      8
#define CDIM    64
#define NTOK    4096
#define HIDDEN  256
#define LPAD    72   // padded LDS row (64 data + 8 pad) to spread banks

#if defined(__has_builtin)
#if __has_builtin(__builtin_amdgcn_global_load_async_to_lds_b128)
#define HAVE_ASYNC_LDS 1
#endif
#if __has_builtin(__builtin_amdgcn_s_wait_asynccnt)
#define HAVE_WAIT_ASYNC 1
#endif
#endif
#ifndef HAVE_ASYNC_LDS
#define HAVE_ASYNC_LDS 0
#endif
#ifndef HAVE_WAIT_ASYNC
#define HAVE_WAIT_ASYNC 0
#endif

static __device__ inline float fsigmoid(float v) {
    return 1.0f / (1.0f + __expf(-v));
}

// B-fragment: 8 bf16 at p and 8 bf16 at p+16 along K (works for LDS or global pointers)
static __device__ inline v16bf load_b_frag(const bf16_t* p) {
    v8bf lo = *(const v8bf*)p;
    v8bf hi = *(const v8bf*)(p + 16);
    v16bf r;
#pragma unroll
    for (int i = 0; i < 8; ++i) { r[i] = lo[i]; r[8 + i] = hi[i]; }
    return r;
}

// A-fragment from an f32 weight row: ISA 16x32 bf16 A-layout
// lanes 0-15: K = c0+0..7 & c0+16..23 ; lanes 16-31: K = c0+8..15 & c0+24..31
static __device__ inline v16bf a_frag_from_w(const float* Wrow, int c0, int lhalf) {
    const float* p0 = Wrow + c0 + lhalf * 8;
    const float* p1 = Wrow + c0 + 16 + lhalf * 8;
    float4 a = *(const float4*)p0, b = *(const float4*)(p0 + 4);
    float4 c = *(const float4*)p1, d = *(const float4*)(p1 + 4);
    float t[16] = {a.x,a.y,a.z,a.w,b.x,b.y,b.z,b.w,
                   c.x,c.y,c.z,c.w,d.x,d.y,d.z,d.w};
    v16bf r;
#pragma unroll
    for (int i = 0; i < 16; ++i) r[i] = (bf16_t)t[i];
    return r;
}

// ---------------------------------------------------------------------------
// Kernel 1: WMMA projections. Per block: 1 batch x 256 positions.
// x, masked-x staged as bf16 [n][c] in LDS (B operands). Weight A-fragments
// converted f32->bf16 on the fly. Outputs: Q,K bf16 [b][n][c]; V bf16
// [b][c][n]; Gate f32 [b][n][c]; GS f32 [b][n].
// ---------------------------------------------------------------------------
__global__ __launch_bounds__(256) void proj_kernel(
    const float* __restrict__ x,  const float* __restrict__ mask,
    const float* __restrict__ Wq, const float* __restrict__ bq,
    const float* __restrict__ Wk, const float* __restrict__ bk,
    const float* __restrict__ Wv, const float* __restrict__ bv,
    const float* __restrict__ Wg, const float* __restrict__ bg,
    const float* __restrict__ W1, const float* __restrict__ b1,
    const float* __restrict__ W2, const float* __restrict__ b2,
    bf16_t* __restrict__ Qo, bf16_t* __restrict__ Ko,
    bf16_t* __restrict__ Vo, float* __restrict__ Gate,
    float* __restrict__ GS)
{
    extern __shared__ char smem[];
    bf16_t* xbf  = (bf16_t*)smem;          // [256][LPAD]
    bf16_t* xmbf = xbf + 256 * LPAD;       // [256][LPAD]

    const int tid   = threadIdx.x;
    const int wave  = tid >> 5;
    const int lane  = tid & 31;
    const int lhalf = lane >> 4;
    const int l16   = lane & 15;
    const int b     = blockIdx.x >> 4;
    const int n0    = (blockIdx.x & 15) << 8;

    // ---- stage x / masked-x to LDS as bf16, transposed to [n][c] ----
    {
        const float  m  = mask[(size_t)b * NTOK + n0 + tid];
        const float* xp = x + (size_t)b * CDIM * NTOK + n0 + tid;
        bf16_t* xr  = xbf  + tid * LPAD;
        bf16_t* xmr = xmbf + tid * LPAD;
#pragma unroll
        for (int c = 0; c < CDIM; ++c) {
            float v = xp[(size_t)c * NTOK];
            xr[c]  = (bf16_t)v;
            xmr[c] = (bf16_t)(v * m);
        }
    }
    __syncthreads();

    // ---- preload B fragments for this wave's two n-tiles ----
    const int nt0 = wave * 2;
    v16bf bx[2][2], bxm[2][2];
#pragma unroll
    for (int t = 0; t < 2; ++t) {
        const int nl = (nt0 + t) * 16 + l16;
#pragma unroll
        for (int ch = 0; ch < 2; ++ch) {
            const int c0 = ch * 32 + lhalf * 8;
            bx[t][ch]  = load_b_frag(xbf  + nl * LPAD + c0);
            bxm[t][ch] = load_b_frag(xmbf + nl * LPAD + c0);
        }
    }

    const float* Wm[4]   = {Wq, Wk, Wv, Wg};
    const float* bias[4] = {bq, bk, bv, bg};

#pragma unroll
    for (int mt = 0; mt < 4; ++mt) {
#pragma unroll 1
        for (int ot = 0; ot < 4; ++ot) {
            const float* wrow = Wm[mt] + (ot * 16 + l16) * CDIM;
            v16bf a0 = a_frag_from_w(wrow, 0, lhalf);
            v16bf a1 = a_frag_from_w(wrow, 32, lhalf);
            float4 blo = *(const float4*)(bias[mt] + ot * 16 + 8 * lhalf);
            float4 bhi = *(const float4*)(bias[mt] + ot * 16 + 8 * lhalf + 4);
            v8f binit = (v8f){blo.x, blo.y, blo.z, blo.w, bhi.x, bhi.y, bhi.z, bhi.w};
#pragma unroll
            for (int t = 0; t < 2; ++t) {
                v16bf b0 = (mt < 2) ? bxm[t][0] : bx[t][0];
                v16bf b1f = (mt < 2) ? bxm[t][1] : bx[t][1];
                v8f acc = binit;
                acc = __builtin_amdgcn_wmma_f32_16x16x32_bf16(
                        false, a0, false, b0, (short)0, acc, false, false);
                acc = __builtin_amdgcn_wmma_f32_16x16x32_bf16(
                        false, a1, false, b1f, (short)0, acc, false, false);
                const int n = n0 + (nt0 + t) * 16 + l16;
                if (mt == 0 || mt == 1) {
                    // Q/K: [b][n][c], rows r are 8 consecutive channels
                    v8bf pk;
#pragma unroll
                    for (int r = 0; r < 8; ++r) pk[r] = (bf16_t)acc[r];
                    bf16_t* dst = (mt == 0 ? Qo : Ko) +
                                  ((size_t)b * NTOK + n) * CDIM + ot * 16 + 8 * lhalf;
                    *(v8bf*)dst = pk;
                } else if (mt == 2) {
                    // V: [b][c][n]
#pragma unroll
                    for (int r = 0; r < 8; ++r) {
                        const int c = ot * 16 + r + 8 * lhalf;
                        Vo[((size_t)b * CDIM + c) * NTOK + n] = (bf16_t)acc[r];
                    }
                } else {
                    // Gate: sigmoid, f32 [b][n][c]
                    float* dst = Gate + ((size_t)b * NTOK + n) * CDIM + ot * 16 + 8 * lhalf;
                    float4 g0 = {fsigmoid(acc[0]), fsigmoid(acc[1]),
                                 fsigmoid(acc[2]), fsigmoid(acc[3])};
                    float4 g1 = {fsigmoid(acc[4]), fsigmoid(acc[5]),
                                 fsigmoid(acc[6]), fsigmoid(acc[7])};
                    *(float4*)dst = g0;
                    *(float4*)(dst + 4) = g1;
                }
            }
        }
    }

    // ---- hidden MLP + gate_strength via WMMA ----
    float gst[2] = {0.0f, 0.0f};
#pragma unroll 1
    for (int ht = 0; ht < 16; ++ht) {
        const float* wrow = W1 + (ht * 16 + l16) * CDIM;
        v16bf a0 = a_frag_from_w(wrow, 0, lhalf);
        v16bf a1 = a_frag_from_w(wrow, 32, lhalf);
        float4 blo = *(const float4*)(b1 + ht * 16 + 8 * lhalf);
        float4 bhi = *(const float4*)(b1 + ht * 16 + 8 * lhalf + 4);
        float4 wlo = *(const float4*)(W2 + ht * 16 + 8 * lhalf);
        float4 whi = *(const float4*)(W2 + ht * 16 + 8 * lhalf + 4);
        v8f binit = (v8f){blo.x, blo.y, blo.z, blo.w, bhi.x, bhi.y, bhi.z, bhi.w};
        float w2v[8] = {wlo.x, wlo.y, wlo.z, wlo.w, whi.x, whi.y, whi.z, whi.w};
#pragma unroll
        for (int t = 0; t < 2; ++t) {
            v8f acc = binit;
            acc = __builtin_amdgcn_wmma_f32_16x16x32_bf16(
                    false, a0, false, bx[t][0], (short)0, acc, false, false);
            acc = __builtin_amdgcn_wmma_f32_16x16x32_bf16(
                    false, a1, false, bx[t][1], (short)0, acc, false, false);
            float part = 0.0f;
#pragma unroll
            for (int r = 0; r < 8; ++r) part += w2v[r] * fmaxf(acc[r], 0.0f);
            gst[t] += part;
        }
    }
#pragma unroll
    for (int t = 0; t < 2; ++t) {
        float tot = gst[t] + __shfl_xor(gst[t], 16);
        tot = fsigmoid(tot + b2[0]);
        if (lhalf == 0)
            GS[(size_t)b * NTOK + n0 + (nt0 + t) * 16 + l16] = tot;
    }
}

// ---------------------------------------------------------------------------
// Kernel 2: flash attention, LDS-staged K/V with double buffering
// (async-to-LDS), bf16 WMMA, gated epilogue.
// ---------------------------------------------------------------------------
__global__ __launch_bounds__(256) void attn_kernel(
    const bf16_t* __restrict__ Q, const bf16_t* __restrict__ K,
    const bf16_t* __restrict__ V, const float* __restrict__ Gate,
    const float* __restrict__ GS, const int* __restrict__ flags,
    const float* __restrict__ x,  float* __restrict__ out)
{
    __shared__ bf16_t Kt[2][64 * LPAD];   // [key][c], padded
    __shared__ bf16_t Vt[2][64 * LPAD];   // [c][n],   padded
    __shared__ bf16_t Pbuf[8][16 * LPAD]; // per-wave P transpose buffer

    const int tid   = threadIdx.x;
    const int wave  = tid >> 5;
    const int lane  = tid & 31;
    const int lhalf = lane >> 4;
    const int l16   = lane & 15;

    const int b    = blockIdx.x >> 5;
    const int qblk = blockIdx.x & 31;
    const int m0   = qblk * 128 + wave * 16;

    const bf16_t* Qb = Q + (size_t)b * NTOK * CDIM;
    const bf16_t* Kb = K + (size_t)b * NTOK * CDIM;
    const bf16_t* Vb = V + (size_t)b * CDIM * NTOK;

    // cooperative staging coords: 64 rows x 4 chunks of 16 elements
    const int srow = tid >> 2;
    const int scol = (tid & 3) << 4;

    auto stage = [&](int bi, int kb) {
        const bf16_t* gk = Kb + (size_t)(kb + srow) * CDIM + scol;
        const bf16_t* gv = Vb + (size_t)srow * NTOK + kb + scol;
        bf16_t* lk = &Kt[bi][srow * LPAD + scol];
        bf16_t* lv = &Vt[bi][srow * LPAD + scol];
#if HAVE_ASYNC_LDS
        __builtin_amdgcn_global_load_async_to_lds_b128(
            (gv4i_t*)(bf16_t*)gk,       (lv4i_t*)lk,       0, 0);
        __builtin_amdgcn_global_load_async_to_lds_b128(
            (gv4i_t*)(bf16_t*)(gk + 8), (lv4i_t*)(lk + 8), 0, 0);
        __builtin_amdgcn_global_load_async_to_lds_b128(
            (gv4i_t*)(bf16_t*)gv,       (lv4i_t*)lv,       0, 0);
        __builtin_amdgcn_global_load_async_to_lds_b128(
            (gv4i_t*)(bf16_t*)(gv + 8), (lv4i_t*)(lv + 8), 0, 0);
#else
        uint4 a0 = *(const uint4*)gk, a1 = *(const uint4*)(gk + 8);
        uint4 c0 = *(const uint4*)gv, c1 = *(const uint4*)(gv + 8);
        *(uint4*)lk = a0; *(uint4*)(lk + 8) = a1;
        *(uint4*)lv = c0; *(uint4*)(lv + 8) = c1;
#endif
    };
    auto wait_stage = [&]() {
#if HAVE_ASYNC_LDS
#if HAVE_WAIT_ASYNC
        __builtin_amdgcn_s_wait_asynccnt(0);
#else
        asm volatile("s_wait_asynccnt 0" ::: "memory");
#endif
#endif
        __syncthreads();
    };

    // ---- A-fragments of Q (rows m0..m0+15) ----
    v16bf qa[2];
    {
        const bf16_t* qrow = Qb + (size_t)(m0 + l16) * CDIM;
#pragma unroll
        for (int ch = 0; ch < 2; ++ch) {
            const int c0 = ch * 32;
            v8bf lo = *(const v8bf*)(qrow + c0 +      lhalf * 8);
            v8bf hi = *(const v8bf*)(qrow + c0 + 16 + lhalf * 8);
#pragma unroll
            for (int i = 0; i < 8; ++i) { qa[ch][i] = lo[i]; qa[ch][8 + i] = hi[i]; }
        }
    }

    v8f O[4];
#pragma unroll
    for (int t = 0; t < 4; ++t) O[t] = (v8f){0.f,0.f,0.f,0.f,0.f,0.f,0.f,0.f};
    float rmax[8], rsum[8];
#pragma unroll
    for (int r = 0; r < 8; ++r) { rmax[r] = -1e30f; rsum[r] = 0.0f; }

    bf16_t* Pw = &Pbuf[wave][0];

    stage(0, 0);
    wait_stage();

    int cur = 0;
    for (int kb = 0; kb < NTOK; kb += 64, cur ^= 1) {
        if (kb + 64 < NTOK) stage(cur ^ 1, kb + 64);

        const bf16_t* Kc = &Kt[cur][0];
        const bf16_t* Vc = &Vt[cur][0];

        // ---------------- S = Q * K^T ----------------
        v8f S[4];
#pragma unroll
        for (int jt = 0; jt < 4; ++jt) {
            v8f acc = (v8f){0.f,0.f,0.f,0.f,0.f,0.f,0.f,0.f};
            const int key = jt * 16 + l16;
#pragma unroll
            for (int ch = 0; ch < 2; ++ch) {
                v16bf bb = load_b_frag(Kc + key * LPAD + ch * 32 + lhalf * 8);
                acc = __builtin_amdgcn_wmma_f32_16x16x32_bf16(
                        false, qa[ch], false, bb, (short)0, acc, false, false);
            }
            S[jt] = acc;
        }

        // ---------------- online softmax ----------------
        float newmax[8];
#pragma unroll
        for (int r = 0; r < 8; ++r) {
            float mx = fmaxf(fmaxf(S[0][r], S[1][r]), fmaxf(S[2][r], S[3][r]));
#pragma unroll
            for (int off = 1; off < 16; off <<= 1)
                mx = fmaxf(mx, __shfl_xor(mx, off));
            newmax[r] = fmaxf(rmax[r], mx);
        }
#pragma unroll
        for (int r = 0; r < 8; ++r) {
            float alpha = __expf(rmax[r] - newmax[r]);
            rsum[r] *= alpha;
#pragma unroll
            for (int t = 0; t < 4; ++t) O[t][r] *= alpha;
            rmax[r] = newmax[r];
        }

        float psum[8];
#pragma unroll
        for (int r = 0; r < 8; ++r) psum[r] = 0.0f;
#pragma unroll
        for (int jt = 0; jt < 4; ++jt) {
#pragma unroll
            for (int r = 0; r < 8; ++r) {
                float p = __expf(S[jt][r] - rmax[r]);
                psum[r] += p;
                Pw[(r + 8 * lhalf) * LPAD + jt * 16 + l16] = (bf16_t)p;
            }
        }
#pragma unroll
        for (int r = 0; r < 8; ++r) {
            float ps = psum[r];
#pragma unroll
            for (int off = 1; off < 16; off <<= 1) ps += __shfl_xor(ps, off);
            rsum[r] += ps;
        }

        // wave-local LDS RAW fence for Pbuf
        asm volatile("s_wait_dscnt 0" ::: "memory");

        // ---------------- read P back as A-fragments ----------------
        v16bf pa[2];
        {
            const bf16_t* prow = Pw + l16 * LPAD;
#pragma unroll
            for (int ch = 0; ch < 2; ++ch) {
                const int k0 = ch * 32;
                v8bf lo = *(const v8bf*)(prow + k0 +      lhalf * 8);
                v8bf hi = *(const v8bf*)(prow + k0 + 16 + lhalf * 8);
#pragma unroll
                for (int i = 0; i < 8; ++i) { pa[ch][i] = lo[i]; pa[ch][8 + i] = hi[i]; }
            }
        }

        // ---------------- O += P * V^T ----------------
#pragma unroll
        for (int ct = 0; ct < 4; ++ct) {
            v8f acc = O[ct];
            const int c = ct * 16 + l16;
#pragma unroll
            for (int ch = 0; ch < 2; ++ch) {
                v16bf bb = load_b_frag(Vc + c * LPAD + ch * 32 + lhalf * 8);
                acc = __builtin_amdgcn_wmma_f32_16x16x32_bf16(
                        false, pa[ch], false, bb, (short)0, acc, false, false);
            }
            O[ct] = acc;
        }

        // staged buffer ready + all waves done with `cur`
        wait_stage();
    }

    // ---------------- epilogue: normalize, gate, select ----------------
    const int    flag = flags[b];
    const float* Gb   = Gate + (size_t)b * NTOK * CDIM;
    const float* GSb  = GS   + (size_t)b * NTOK;
    const float* xb   = x    + (size_t)b * CDIM * NTOK;
    float*       ob   = out  + (size_t)b * CDIM * NTOK;

#pragma unroll
    for (int r = 0; r < 8; ++r) {
        const int   m   = m0 + r + 8 * lhalf;
        const float inv = 1.0f / rsum[r];
        const float gsv = GSb[m];
#pragma unroll
        for (int ct = 0; ct < 4; ++ct) {
            const int c = ct * 16 + l16;
            float res;
            if (flag)
                res = O[ct][r] * inv * Gb[(size_t)m * CDIM + c] * gsv;
            else
                res = xb[(size_t)c * NTOK + m];
            ob[(size_t)c * NTOK + m] = res;
        }
    }
}

// ---------------------------------------------------------------------------
extern "C" void kernel_launch(void* const* d_in, const int* in_sizes, int n_in,
                              void* d_out, int out_size, void* d_ws, size_t ws_size,
                              hipStream_t stream) {
    const float* x    = (const float*)d_in[0];
    const float* mask = (const float*)d_in[1];
    const int*   flg  = (const int*)  d_in[2];
    const float* Wq = (const float*)d_in[3];  const float* bq = (const float*)d_in[4];
    const float* Wk = (const float*)d_in[5];  const float* bk = (const float*)d_in[6];
    const float* Wv = (const float*)d_in[7];  const float* bv = (const float*)d_in[8];
    const float* Wg = (const float*)d_in[9];  const float* bg = (const float*)d_in[10];
    const float* W1 = (const float*)d_in[11]; const float* b1 = (const float*)d_in[12];
    const float* W2 = (const float*)d_in[13]; const float* b2 = (const float*)d_in[14];

    char* ws = (char*)d_ws;
    const size_t nqk = (size_t)NB * NTOK * CDIM;       // 2 Mi elements
    bf16_t* Qo   = (bf16_t*)(ws);                      // 4 MB
    bf16_t* Ko   = (bf16_t*)(ws + nqk * 2);            // 4 MB
    bf16_t* Vo   = (bf16_t*)(ws + nqk * 4);            // 4 MB
    float*  Gate = (float*) (ws + nqk * 6);            // 8 MB
    float*  GS   = (float*) (ws + nqk * 6 + nqk * 4);  // 128 KB

    const size_t proj_lds = 2u * 256u * LPAD * sizeof(bf16_t); // 73,728 B

    proj_kernel<<<NB * (NTOK / 256), 256, proj_lds, stream>>>(
        x, mask, Wq, bq, Wk, bk, Wv, bv, Wg, bg, W1, b1, W2, b2,
        Qo, Ko, Vo, Gate, GS);

    attn_kernel<<<NB * (NTOK / 128), 256, 0, stream>>>(
        Qo, Ko, Vo, Gate, GS, flg, x, (float*)d_out);
}
